// GraphAttentionLayer_88897233092941
// MI455X (gfx1250) — matrix operand order repaired and checked by reading, not compiled, not run on hardware
//
#include <hip/hip_runtime.h>
#include <hip/hip_bf16.h>

// GAT layer: N=4096, F_IN=512, H=4, D=64. Flash-style attention with
// v_wmma_f32_16x16x32_f16 for both GEMMs on gfx1250 (wave32).

#define N_NODES 4096
#define F_IN    512
#define N_HEADS 4
#define HEAD_D  64
#define OUT_F   (N_HEADS * HEAD_D)   // 256

typedef _Float16 v16h __attribute__((ext_vector_type(16)));
typedef _Float16 v8h  __attribute__((ext_vector_type(8)));
typedef float    v8f  __attribute__((ext_vector_type(8)));

static __device__ __forceinline__ float leaky02(float x) {
    // slope 0.2: for x<0, 0.2x > x; for x>=0, x >= 0.2x
    return fmaxf(x, 0.2f * x);
}

// Load an f16 A-fragment (16x32, row-major source, leading dim = ld halves).
// rowptr points at the start of this lane's row.
static __device__ __forceinline__ v16h load_a_frag(const _Float16* rowptr, int k0, int lane) {
    int koff = (lane >> 4) << 4;                      // 0 or 16
    const _Float16* p = rowptr + k0 + koff;
    v8h lo = *(const v8h*)(p);                        // K = k0+koff .. +7
    v8h hi = *(const v8h*)(p + 8);                    // K = k0+koff+8 .. +15
    v16h a;
#pragma unroll
    for (int q = 0; q < 8; ++q) { a[q] = lo[q]; a[8 + q] = hi[q]; }
    return a;
}

// ---------------------------------------------------------------------------
// Kernel 0: convert x -> f16 row-major; W[h][f][d] -> Wt f16 [h][d][f]
// ---------------------------------------------------------------------------
__global__ __launch_bounds__(256)
void cvt_kernel(const float* __restrict__ x, const float* __restrict__ W,
                _Float16* __restrict__ xh, _Float16* __restrict__ wt) {
    int idx = blockIdx.x * 256 + threadIdx.x;
    const int total_x = N_NODES * F_IN;          // 2097152
    const int total_w = N_HEADS * F_IN * HEAD_D; // 131072
    if (idx < total_x) xh[idx] = (_Float16)x[idx];
    if (idx < total_w) {
        int h = idx >> 15;            // / (512*64)
        int rem = idx & 32767;
        int f = rem >> 6;
        int d = rem & 63;
        wt[((h * HEAD_D + d) * F_IN) + f] = (_Float16)W[idx];
    }
}

// ---------------------------------------------------------------------------
// Kernel 1: WxT[h][d][n] = sum_f x[n][f] * W[h][f][d], f16 out, WMMA core.
// One wave per (16-row tile, head): 4 d-tiles of 16, K loop 512/32.
// ---------------------------------------------------------------------------
__global__ __launch_bounds__(32)
void wx_kernel(const _Float16* __restrict__ xh, const _Float16* __restrict__ wt,
               _Float16* __restrict__ wxT) {
    int itile = blockIdx.x;        // 0..255
    int h     = blockIdx.y;        // 0..3
    int lane  = threadIdx.x;       // 0..31
    int col   = lane & 15;
    int kseg  = (lane >> 4) << 4;  // 0 or 16 (B fragment K segment)

    const _Float16* xrow = xh + (size_t)(itile * 16 + col) * F_IN;

    v8f c[4] = {v8f{}, v8f{}, v8f{}, v8f{}};

#pragma unroll 4
    for (int k0 = 0; k0 < F_IN; k0 += 32) {
        v16h a = load_a_frag(xrow, k0, lane);
#pragma unroll
        for (int t = 0; t < 4; ++t) {
            const _Float16* bp =
                wt + (size_t)(h * HEAD_D + t * 16 + col) * F_IN + k0 + kseg;
            v16h b = *(const v16h*)bp;  // contiguous 32B (two b128 loads)
            c[t] = __builtin_amdgcn_wmma_f32_16x16x32_f16(
                false, a, false, b, (short)0, c[t], false, false);
        }
    }

    // C layout: lane holds col=lane&15, rows rbase..rbase+7 (consecutive n)
    int nbase = itile * 16 + ((lane >> 4) << 3);
#pragma unroll
    for (int t = 0; t < 4; ++t) {
        int d = t * 16 + col;
        v8h o;
#pragma unroll
        for (int r = 0; r < 8; ++r) o[r] = (_Float16)c[t][r];
        *(v8h*)(wxT + (size_t)(h * HEAD_D + d) * N_NODES + nbase) = o;
    }
}

// ---------------------------------------------------------------------------
// Kernel 2: s_src[h][n], s_dst[h][n], Mh[h] = max_n s_dst[h][n]
// ---------------------------------------------------------------------------
__global__ __launch_bounds__(256)
void s_kernel(const _Float16* __restrict__ wxT, const float* __restrict__ a,
              float* __restrict__ ssrc, float* __restrict__ sdst,
              float* __restrict__ Mh) {
    int h = blockIdx.x;
    int t = threadIdx.x;
    __shared__ float red[256];
    float localmax = -3.0e38f;
    for (int n = t; n < N_NODES; n += 256) {
        float as = 0.0f, ad = 0.0f;
#pragma unroll 8
        for (int d = 0; d < HEAD_D; ++d) {
            float w = (float)wxT[(size_t)(h * HEAD_D + d) * N_NODES + n];
            as += a[h * (2 * HEAD_D) + d] * w;
            ad += a[h * (2 * HEAD_D) + HEAD_D + d] * w;
        }
        ssrc[h * N_NODES + n] = as;
        sdst[h * N_NODES + n] = ad;
        localmax = fmaxf(localmax, ad);
    }
    red[t] = localmax;
    __syncthreads();
    for (int s = 128; s > 0; s >>= 1) {
        if (t < s) red[t] = fmaxf(red[t], red[t + s]);
        __syncthreads();
    }
    if (t == 0) Mh[h] = red[0];
}

// ---------------------------------------------------------------------------
// Kernel 3: flash-style masked softmax + att x Wx, WMMA core.
// 256 blocks (16-row tiles) x 4 waves (wave = head). One-pass softmax using
// mhat_i = leaky(s_src[i] + max_j s_dst[j]) >= true row max (leaky monotone).
// ---------------------------------------------------------------------------
__global__ __launch_bounds__(128)
void att_kernel(const int* __restrict__ adj, const _Float16* __restrict__ wxT,
                const float* __restrict__ ssrc, const float* __restrict__ sdst,
                const float* __restrict__ Mh, const float* __restrict__ bias,
                float* __restrict__ out) {
    int itile = blockIdx.x;
    int h     = threadIdx.x >> 5;   // wave id == head
    int lane  = threadIdx.x & 31;
    int row   = lane & 15;
    int i     = itile * 16 + row;
    int koff  = (lane >> 4) << 4;   // A-fragment K offset: 0 or 16
    int kseg  = (lane >> 4) << 4;   // B-fragment K segment: 0 or 16

    float my_ssrc = ssrc[h * N_NODES + i];
    float mhat    = leaky02(my_ssrc + Mh[h]);

    const int*   adjrow = adj + (size_t)i * N_NODES;
    const float* sd     = sdst + h * N_NODES;

    v8f c[4] = {v8f{}, v8f{}, v8f{}, v8f{}};
    float lsum = 0.0f;

    for (int j0 = 0; j0 < N_NODES; j0 += 32) {
        __builtin_prefetch(adjrow + j0 + 2048, 0, 1);

        int4 ma0 = *(const int4*)(adjrow + j0 + koff);
        int4 ma1 = *(const int4*)(adjrow + j0 + koff + 4);
        int4 mb0 = *(const int4*)(adjrow + j0 + koff + 8);
        int4 mb1 = *(const int4*)(adjrow + j0 + koff + 12);
        float4 sa0 = *(const float4*)(sd + j0 + koff);
        float4 sa1 = *(const float4*)(sd + j0 + koff + 4);
        float4 sb0 = *(const float4*)(sd + j0 + koff + 8);
        float4 sb1 = *(const float4*)(sd + j0 + koff + 12);

        int   am[16] = {ma0.x, ma0.y, ma0.z, ma0.w, ma1.x, ma1.y, ma1.z, ma1.w,
                        mb0.x, mb0.y, mb0.z, mb0.w, mb1.x, mb1.y, mb1.z, mb1.w};
        float sv[16] = {sa0.x, sa0.y, sa0.z, sa0.w, sa1.x, sa1.y, sa1.z, sa1.w,
                        sb0.x, sb0.y, sb0.z, sb0.w, sb1.x, sb1.y, sb1.z, sb1.w};

        v16h a;
#pragma unroll
        for (int q = 0; q < 16; ++q) {
            float e  = leaky02(my_ssrc + sv[q]);
            float pq = (am[q] > 0) ? __expf(e - mhat) : 0.0f;
            _Float16 ph = (_Float16)pq;
            a[q] = ph;
            lsum += (float)ph;
        }

#pragma unroll
        for (int t = 0; t < 4; ++t) {
            const _Float16* bp =
                wxT + (size_t)(h * HEAD_D + t * 16 + row) * N_NODES + j0 + kseg;
            v16h b = *(const v16h*)bp;
            c[t] = __builtin_amdgcn_wmma_f32_16x16x32_f16(
                false, a, false, b, (short)0, c[t], false, false);
        }
    }

    // Row sums: lanes L and L+16 hold disjoint halves of row L&15.
    float tot = lsum + __shfl_xor(lsum, 16, 32);
    __shared__ float lrow[N_HEADS][16];
    if (lane < 16) lrow[h][lane] = tot;
    __syncthreads();

    // C layout: lane holds col = 16t + (lane&15), rows rbase..rbase+7
    int rbase = (lane >> 4) << 3;
#pragma unroll
    for (int t = 0; t < 4; ++t) {
        int d = t * 16 + row;
        float bi = bias[h * HEAD_D + d];
#pragma unroll
        for (int r = 0; r < 8; ++r) {
            float l = fmaxf(lrow[h][rbase + r], 1e-30f);
            float v = c[t][r] / l + bi;
            v = fmaxf(v, 0.0f);
            out[(size_t)(itile * 16 + rbase + r) * OUT_F + h * HEAD_D + d] = v;
        }
    }
}

// ---------------------------------------------------------------------------
extern "C" void kernel_launch(void* const* d_in, const int* in_sizes, int n_in,
                              void* d_out, int out_size, void* d_ws, size_t ws_size,
                              hipStream_t stream) {
    const float* x    = (const float*)d_in[0];   // [4096,512]
    const int*   adj  = (const int*)d_in[1];     // [4096,4096]
    const float* W    = (const float*)d_in[2];   // [4,512,64]
    const float* a    = (const float*)d_in[3];   // [4,128]
    const float* bias = (const float*)d_in[4];   // [256]
    float* out = (float*)d_out;                  // [4096,256]

    char* ws = (char*)d_ws;
    _Float16* xh   = (_Float16*)(ws);                 // 4096*512*2   = 4194304 B
    _Float16* wt   = (_Float16*)(ws + 4194304);       // 4*64*512*2   =  262144 B
    _Float16* wxT  = (_Float16*)(ws + 4456448);       // 4*64*4096*2  = 2097152 B
    float*    ssrc = (float*)(ws + 6553600);          // 4*4096*4     =   65536 B
    float*    sdst = (float*)(ws + 6619136);          // 4*4096*4     =   65536 B
    float*    Mh   = (float*)(ws + 6684672);          // 16 B

    // 1) fp32 -> f16 conversion (+ W transpose for B-fragment loads)
    cvt_kernel<<<(N_NODES * F_IN + 255) / 256, 256, 0, stream>>>(x, W, xh, wt);

    // 2) WxT = (x @ W[h])^T per head, WMMA
    wx_kernel<<<dim3(N_NODES / 16, N_HEADS), 32, 0, stream>>>(xh, wt, wxT);

    // 3) attention logits sources + per-head global max of s_dst
    s_kernel<<<N_HEADS, 256, 0, stream>>>(wxT, a, ssrc, sdst, Mh);

    // 4) masked softmax + att @ Wx, WMMA, fused bias + relu
    att_kernel<<<N_NODES / 16, 128, 0, stream>>>(adj, wxT, ssrc, sdst, Mh, bias, out);
}
